// RelativePositionAttention_3367254360520
// MI455X (gfx1250) — compile-verified, matrix operand order checked by reading, not run on hardware
//
#include <hip/hip_runtime.h>
#include <hip/hip_bf16.h>

#define BATCH   8
#define SEQ     512
#define EMB     768
#define HEADS   12
#define HDIM    64
#define MAXLEN  512

typedef __bf16 bf16_t;
typedef __attribute__((ext_vector_type(8)))  bf16_t v8bf;
typedef __attribute__((ext_vector_type(16))) bf16_t v16bf;
typedef __attribute__((ext_vector_type(8)))  float  v8f;
typedef __attribute__((ext_vector_type(4)))  float  v4f;

static __device__ __forceinline__ bf16_t f2bf(float f) {
  unsigned u = __builtin_bit_cast(unsigned, f);
  unsigned r = u + 0x7FFFu + ((u >> 16) & 1u);      // round-to-nearest-even
  unsigned short h = (unsigned short)(r >> 16);
  return __builtin_bit_cast(bf16_t, h);
}

static __device__ __forceinline__ v16bf cat8(v8bf lo, v8bf hi) {
  return __builtin_shufflevector(lo, hi, 0,1,2,3,4,5,6,7,8,9,10,11,12,13,14,15);
}

// Load one 16-bf16 WMMA operand (A or B layout) from a row-major row.
// element e -> column  kbase + (e>>3)*16 + half*8 + (e&7)  (two 16B reads/lane)
static __device__ __forceinline__ v16bf load_tile_row(const bf16_t* row, int kbase, int half) {
  v8bf lo = *(const v8bf*)(row + kbase + half * 8);
  v8bf hi = *(const v8bf*)(row + kbase + 16 + half * 8);
  return cat8(lo, hi);
}

static __device__ __forceinline__ v8f wmma_bf16(v16bf a, v16bf b, v8f c) {
  return __builtin_amdgcn_wmma_f32_16x16x32_bf16(false, a, false, b, (short)0, c, false, false);
}

static __device__ __forceinline__ int clampi(int x, int lo, int hi) {
  return x < lo ? lo : (x > hi ? hi : x);
}

// ----- CDNA5 async Global->LDS copy (ASYNCcnt path, CDNA5 ISA)
// LDS dst byte offset = low 32 bits of generic pointer (ISA §10.2 aperture rule).
static __device__ __forceinline__ unsigned lds_off(const void* p) {
  return (unsigned)(size_t)p;
}
static __device__ __forceinline__ void async_copy_b128(unsigned ldsdst, const bf16_t* src) {
  asm volatile("global_load_async_to_lds_b128 %0, %1, off"
               :: "v"(ldsdst), "v"(src) : "memory");
}
static __device__ __forceinline__ void wait_async0() {
  asm volatile("s_wait_asynccnt 0x0" ::: "memory");
}

// ---------------------------------------------------------------- prep kernels
// 8 floats -> 8 bf16 per thread (b128+b128 in, b128 out)
__global__ void cvt_f32_bf16(const float* __restrict__ in, bf16_t* __restrict__ out, int n8) {
  int i = blockIdx.x * 256 + threadIdx.x;
  if (i >= n8) return;
  v4f a = ((const v4f*)in)[2 * i];
  v4f b = ((const v4f*)in)[2 * i + 1];
  v8bf o;
#pragma unroll
  for (int j = 0; j < 4; ++j) { o[j] = f2bf(a[j]); o[4 + j] = f2bf(b[j]); }
  ((v8bf*)out)[i] = o;
}

// Wt[n*E + k] = W[k*E + n]  via LDS 32x32 tile (coalesced both directions)
__global__ __launch_bounds__(256)
void transpose_cvt(const float* __restrict__ W, bf16_t* __restrict__ Wt) {
  __shared__ bf16_t tile[32][33];
  const int tx = threadIdx.x & 31;
  const int ty = threadIdx.x >> 5;             // 0..7
  const int bx = blockIdx.x % (EMB / 32);
  const int by = blockIdx.x / (EMB / 32);
  const int x  = bx * 32 + tx;                 // n  (col of W)
#pragma unroll
  for (int j = 0; j < 32; j += 8) {
    const int y = by * 32 + ty + j;            // k  (row of W)
    tile[ty + j][tx] = f2bf(W[(size_t)y * EMB + x]);
  }
  __syncthreads();
#pragma unroll
  for (int j = 0; j < 32; j += 8) {
    const int n = bx * 32 + ty + j;
    const int k = by * 32 + tx;
    Wt[(size_t)n * EMB + k] = tile[tx][ty + j];
  }
}

// ---------------------------------------------------------------- WMMA GEMM
// Y[m,n] = sum_k A[m,k]*Bt[n,k].  4 waves/block, wave tile 32x64 (8 WMMA / 12 loads per k-step).
// mode 0: bf16 row-major. mode 1: bf16 V-transposed [b,h,d,s]. mode 2: f32 + bias.
__global__ __launch_bounds__(128)
void wmma_gemm_bf16(const bf16_t* __restrict__ A, const bf16_t* __restrict__ Bt,
                    bf16_t* __restrict__ outB, float* __restrict__ outF,
                    const float* __restrict__ bias, int N, int K, int mode)
{
  const int wave = threadIdx.x >> 5;
  const int lane = threadIdx.x & 31;
  const int half = lane >> 4;
  const int l16  = lane & 15;
  const int m0 = blockIdx.y * 128 + wave * 32;
  const int n0 = blockIdx.x * 64;

  const bf16_t* ar0 = A + (size_t)(m0 + l16) * K;
  const bf16_t* ar1 = A + (size_t)(m0 + 16 + l16) * K;
  const bf16_t* br0 = Bt + (size_t)(n0 +  0 + l16) * K;
  const bf16_t* br1 = Bt + (size_t)(n0 + 16 + l16) * K;
  const bf16_t* br2 = Bt + (size_t)(n0 + 32 + l16) * K;
  const bf16_t* br3 = Bt + (size_t)(n0 + 48 + l16) * K;

  v8f acc[2][4] = {};
  for (int kk = 0; kk < K; kk += 32) {
    v16bf a0 = load_tile_row(ar0, kk, half);
    v16bf a1 = load_tile_row(ar1, kk, half);
    v16bf b0 = load_tile_row(br0, kk, half);
    v16bf b1 = load_tile_row(br1, kk, half);
    v16bf b2 = load_tile_row(br2, kk, half);
    v16bf b3 = load_tile_row(br3, kk, half);
    acc[0][0] = wmma_bf16(a0, b0, acc[0][0]);
    acc[1][0] = wmma_bf16(a1, b0, acc[1][0]);
    acc[0][1] = wmma_bf16(a0, b1, acc[0][1]);
    acc[1][1] = wmma_bf16(a1, b1, acc[1][1]);
    acc[0][2] = wmma_bf16(a0, b2, acc[0][2]);
    acc[1][2] = wmma_bf16(a1, b2, acc[1][2]);
    acc[0][3] = wmma_bf16(a0, b3, acc[0][3]);
    acc[1][3] = wmma_bf16(a1, b3, acc[1][3]);
  }

#pragma unroll
  for (int mt = 0; mt < 2; ++mt) {
#pragma unroll
    for (int t = 0; t < 4; ++t) {
#pragma unroll
      for (int v = 0; v < 8; ++v) {
        const int row = m0 + mt * 16 + v + 8 * half;
        const int col = n0 + t * 16 + l16;
        const float val = acc[mt][t][v];
        if (mode == 0) {
          outB[(size_t)row * N + col] = f2bf(val);
        } else if (mode == 1) {
          const int bb = row >> 9, ss = row & (SEQ - 1);
          const int hh = col >> 6, dd = col & (HDIM - 1);
          outB[((size_t)(bb * HEADS + hh) * HDIM + dd) * SEQ + ss] = f2bf(val);
        } else {
          outF[(size_t)row * N + col] = val + bias[col];
        }
      }
    }
  }
}

// ---------------------------------------------------------------- fused attention
// grid: B*H*(S/64) blocks of 128 threads (4 waves); wave owns 16 query rows.
// K/V tiles double-buffered in LDS via async global->LDS copies (shared by all 4 waves).
__global__ __launch_bounds__(128)
void wmma_attention(const bf16_t* __restrict__ Qb, const bf16_t* __restrict__ Kb,
                    const bf16_t* __restrict__ Vt, const float* __restrict__ rel_pos,
                    const int* __restrict__ mask, bf16_t* __restrict__ attn_out)
{
  __shared__ float lds_rel[1024];                       // rel_pos[:, h]
  __shared__ int   lds_mask[SEQ];                       // mask[b, :]
  __shared__ __align__(16) bf16_t lds_P[4][16][32];     // per-wave P staging
  __shared__ __align__(16) bf16_t kbuf[2][32][64];      // K block  [s][d], double buffered
  __shared__ __align__(16) bf16_t vbuf[2][64][32];      // V block  [d][s], double buffered

  const int qblk = blockIdx.x & 7;
  const int h    = (blockIdx.x >> 3) % HEADS;
  const int b    = blockIdx.x / (8 * HEADS);

  // Issue async stage of K/V block k0 into buffer `buf` (256 16B chunks each,
  // 128 threads -> 2+2 async instructions per wave).
  auto issue_kv = [&](int k0, int buf) {
    for (int c = threadIdx.x; c < 256; c += 128) {
      const int kr = c >> 3, kg = c & 7;                // K: 32 rows x 8 groups of 8 bf16
      async_copy_b128(lds_off(&kbuf[buf][kr][kg * 8]),
                      Kb + (size_t)(b * SEQ + k0 + kr) * EMB + h * HDIM + kg * 8);
      const int vd = c >> 2, vg = c & 3;                // V: 64 rows x 4 groups of 8 bf16
      async_copy_b128(lds_off(&vbuf[buf][vd][vg * 8]),
                      Vt + ((size_t)(b * HEADS + h) * HDIM + vd) * SEQ + k0 + vg * 8);
    }
  };

  issue_kv(0, 0);                                       // prologue stage

  for (int i = threadIdx.x; i < 2 * MAXLEN - 1; i += 128) lds_rel[i]  = rel_pos[i * HEADS + h];
  for (int i = threadIdx.x; i < SEQ;            i += 128) lds_mask[i] = mask[b * SEQ + i];

  const int wave = threadIdx.x >> 5;
  const int lane = threadIdx.x & 31;
  const int half = lane >> 4;
  const int l16  = lane & 15;
  const int q0   = qblk * 64 + wave * 16;

  const bf16_t* qrow = Qb + (size_t)(b * SEQ + q0 + l16) * EMB + h * HDIM;
  const v16bf qa0 = load_tile_row(qrow, 0,  half);
  const v16bf qa1 = load_tile_row(qrow, 32, half);

  float m_r[8], l_r[8];
#pragma unroll
  for (int v = 0; v < 8; ++v) { m_r[v] = -__builtin_inff(); l_r[v] = 0.0f; }
  v8f O[4] = {};

  const float scale  = 0.125f;                          // 1/sqrt(64)
  const float NEGBIG = -3.402823466e38f;

  for (int k0 = 0; k0 < SEQ; k0 += 32) {
    const int cur = (k0 >> 5) & 1;
    wait_async0();                                      // my async stores to LDS landed
    __syncthreads();                                    // everyone's landed; prev buffer free
    if (k0 + 32 < SEQ) issue_kv(k0 + 32, cur ^ 1);      // prefetch next block

    // S[16x32] = Q[16x64] . K^T   (B operand rows from LDS K tile)
    v8f s0 = {}, s1 = {};
    s0 = wmma_bf16(qa0, load_tile_row(&kbuf[cur][l16][0],      0,  half), s0);
    s0 = wmma_bf16(qa1, load_tile_row(&kbuf[cur][l16][0],      32, half), s0);
    s1 = wmma_bf16(qa0, load_tile_row(&kbuf[cur][16 + l16][0], 0,  half), s1);
    s1 = wmma_bf16(qa1, load_tile_row(&kbuf[cur][16 + l16][0], 32, half), s1);

    const int kc0 = k0 + l16, kc1 = k0 + 16 + l16;
    const int mk0 = lds_mask[kc0], mk1 = lds_mask[kc1];

    float alpha[8];
#pragma unroll
    for (int v = 0; v < 8; ++v) {
      const int q  = q0 + v + 8 * half;
      const int mq = lds_mask[q];
      const int i0 = clampi(kc0 - q + (MAXLEN - 1), 0, 2 * MAXLEN - 2);
      const int i1 = clampi(kc1 - q + (MAXLEN - 1), 0, 2 * MAXLEN - 2);
      float x0 = s0[v] * scale + lds_rel[i0];
      float x1 = s1[v] * scale + lds_rel[i1];
      if (mq == 0 || mk0 == 0) x0 = NEGBIG;
      if (mq == 0 || mk1 == 0) x1 = NEGBIG;
      float mx = fmaxf(x0, x1);                         // row reduce in 16-lane half
      mx = fmaxf(mx, __shfl_xor(mx, 1));
      mx = fmaxf(mx, __shfl_xor(mx, 2));
      mx = fmaxf(mx, __shfl_xor(mx, 4));
      mx = fmaxf(mx, __shfl_xor(mx, 8));
      const float mnew = fmaxf(m_r[v], mx);
      const float a    = __expf(m_r[v] - mnew);
      const float p0   = __expf(x0 - mnew);
      const float p1   = __expf(x1 - mnew);
      float rs = p0 + p1;
      rs += __shfl_xor(rs, 1);
      rs += __shfl_xor(rs, 2);
      rs += __shfl_xor(rs, 4);
      rs += __shfl_xor(rs, 8);
      l_r[v] = l_r[v] * a + rs;
      m_r[v] = mnew;
      alpha[v] = a;
      lds_P[wave][v + 8 * half][l16]      = f2bf(p0);   // C->A relayout via LDS
      lds_P[wave][v + 8 * half][16 + l16] = f2bf(p1);
    }

#pragma unroll
    for (int t = 0; t < 4; ++t)
#pragma unroll
      for (int v = 0; v < 8; ++v)
        O[t][v] *= alpha[v];

    const v16bf pa = load_tile_row(&lds_P[wave][l16][0], 0, half);

    // O[16x64] += P[16x32] . V[32x64]  (B operand rows from LDS V tile)
#pragma unroll
    for (int t = 0; t < 4; ++t) {
      const v16bf vb = load_tile_row(&vbuf[cur][t * 16 + l16][0], 0, half);
      O[t] = wmma_bf16(pa, vb, O[t]);
    }
  }

  float inv[8];
#pragma unroll
  for (int v = 0; v < 8; ++v) inv[v] = 1.0f / l_r[v];

#pragma unroll
  for (int t = 0; t < 4; ++t)
#pragma unroll
    for (int v = 0; v < 8; ++v)
      attn_out[(size_t)(b * SEQ + q0 + v + 8 * half) * EMB + h * HDIM + t * 16 + l16] =
          f2bf(O[t][v] * inv[v]);
}

// ---------------------------------------------------------------- launcher
extern "C" void kernel_launch(void* const* d_in, const int* in_sizes, int n_in,
                              void* d_out, int out_size, void* d_ws, size_t ws_size,
                              hipStream_t stream)
{
  (void)in_sizes; (void)n_in; (void)out_size;
  const float* x    = (const float*)d_in[0];
  const int*   mask = (const int*)  d_in[1];
  const float* Wq   = (const float*)d_in[2];
  const float* Wk   = (const float*)d_in[3];
  const float* Wv   = (const float*)d_in[4];
  const float* rel  = (const float*)d_in[5];
  const float* Wo   = (const float*)d_in[6];
  const float* bo   = (const float*)d_in[7];
  float* out = (float*)d_out;

  const size_t ROWS = (size_t)BATCH * SEQ;                 // 4096
  const size_t XSZ  = ROWS * EMB * sizeof(bf16_t);         // 6 MB
  const size_t WSZ  = (size_t)EMB * EMB * sizeof(bf16_t);  // 1.125 MB
  if (ws_size < 5 * XSZ + 4 * WSZ) return;

  char* p = (char*)d_ws;
  bf16_t* xb    = (bf16_t*)p; p += XSZ;
  bf16_t* Qb    = (bf16_t*)p; p += XSZ;
  bf16_t* Kb    = (bf16_t*)p; p += XSZ;
  bf16_t* Vtb   = (bf16_t*)p; p += XSZ;
  bf16_t* attnb = (bf16_t*)p; p += XSZ;
  bf16_t* WqT   = (bf16_t*)p; p += WSZ;
  bf16_t* WkT   = (bf16_t*)p; p += WSZ;
  bf16_t* WvT   = (bf16_t*)p; p += WSZ;
  bf16_t* WoT   = (bf16_t*)p; p += WSZ;

  const int n8 = (int)(ROWS * EMB / 8);
  cvt_f32_bf16<<<(n8 + 255) / 256, 256, 0, stream>>>(x, xb, n8);

  const int tgrid = (EMB / 32) * (EMB / 32);               // 576
  transpose_cvt<<<tgrid, 256, 0, stream>>>(Wq, WqT);
  transpose_cvt<<<tgrid, 256, 0, stream>>>(Wk, WkT);
  transpose_cvt<<<tgrid, 256, 0, stream>>>(Wv, WvT);
  transpose_cvt<<<tgrid, 256, 0, stream>>>(Wo, WoT);

  dim3 ggrid(EMB / 64, (unsigned)(ROWS / 128));            // (12, 32)
  wmma_gemm_bf16<<<ggrid, 128, 0, stream>>>(xb, WqT, Qb,  nullptr, nullptr, EMB, EMB, 0);
  wmma_gemm_bf16<<<ggrid, 128, 0, stream>>>(xb, WkT, Kb,  nullptr, nullptr, EMB, EMB, 0);
  wmma_gemm_bf16<<<ggrid, 128, 0, stream>>>(xb, WvT, Vtb, nullptr, nullptr, EMB, EMB, 1);

  wmma_attention<<<BATCH * HEADS * (SEQ / 64), 128, 0, stream>>>(Qb, Kb, Vtb, rel, mask, attnb);

  wmma_gemm_bf16<<<ggrid, 128, 0, stream>>>(attnb, WoT, nullptr, out, bo, EMB, EMB, 2);
}
